// MSDN_BASE_65652870087588
// MI455X (gfx1250) — matrix operand order repaired and checked by reading, not compiled, not run on hardware
//
#include <hip/hip_runtime.h>
#include <hip/hip_bf16.h>

typedef __attribute__((ext_vector_type(16))) _Float16 v16h;
typedef __attribute__((ext_vector_type(8)))  _Float16 v8h;
typedef __attribute__((ext_vector_type(8)))  float    v8f;

#define N_NODES 4096
#define N_EDGES 65536
#define FEA     512
#define GATE_W  128
#define TILE_E  128          // edges per block
#define KC      64           // K-chunk (halves)
#define ASTR    72           // LDS row stride in halves (16B-aligned rows, bank-spread)
#define WSTR    72
#define NCHUNK  ((2 * FEA) / KC)   // 16

__device__ __forceinline__ float sigmoidf_fast(float x) {
    return 1.0f / (1.0f + __expf(-x));
}

// ---- CDNA5 async global->LDS copy (ASYNCcnt-tracked), 16B per lane ----
__device__ __forceinline__ void async_load_lds16(const void* g, void* lds) {
    unsigned       l  = (unsigned)(unsigned long long)lds;   // low 32 bits = LDS offset
    unsigned long long ga = (unsigned long long)g;
    asm volatile("global_load_async_to_lds_b128 %0, %1, off"
                 :: "v"(l), "v"(ga) : "memory");
}
__device__ __forceinline__ void wait_async0() {
    asm volatile("s_wait_asynccnt 0x0" ::: "memory");
}

// ---------------------------------------------------------------------------
// Prep: zero sums+counts, pack w (f32 -> f16) chunk-major into ws:
//   wPack[kc][n][j] = (f16) w[n][kc*64 + j]   (exact per-chunk tile image)
// ---------------------------------------------------------------------------
__global__ void __launch_bounds__(256) msdn_prep_kernel(
    const float* __restrict__ w, float* __restrict__ out,
    float* __restrict__ cnt, _Float16* __restrict__ wPack)
{
    int i = blockIdx.x * 256 + threadIdx.x;
    if (i < N_NODES * FEA) out[i] = 0.0f;
    if (i < N_NODES)       cnt[i] = 0.0f;
    if (i < GATE_W * 2 * FEA) {
        int n = i >> 10, k = i & 1023;
        int kc = k >> 6, j = k & 63;
        wPack[(size_t)kc * (GATE_W * KC) + n * KC + j] = (_Float16)w[i];
    }
}

// ---------------------------------------------------------------------------
// Main: gather -> relu -> f16 WMMA GEMM -> sigmoid/mean -> gated scatter-add
// Double-buffered LDS; W tiles via async-to-LDS; A gathers issued pre-compute.
// ---------------------------------------------------------------------------
__global__ void __launch_bounds__(256) msdn_edge_kernel(
    const float*    __restrict__ tgtF,   // [N_NODES, FEA]
    const float*    __restrict__ srcF,   // [N_NODES, FEA]
    const _Float16* __restrict__ wPack,  // [16][128][64] f16
    const float*    __restrict__ bvec,   // [GATE_W]
    const int*      __restrict__ tIdx,
    const int*      __restrict__ sIdx,
    float* __restrict__ out,             // [N_NODES, FEA] (sums)
    float* __restrict__ cnt)             // [N_NODES]
{
    __shared__ __align__(16) _Float16 Ash[2 * TILE_E * ASTR];
    __shared__ __align__(16) _Float16 Wsh[2 * GATE_W * WSTR];
    __shared__ int   tiSh[TILE_E];
    __shared__ int   siSh[TILE_E];
    __shared__ float gateSh[TILE_E];

    const int t    = threadIdx.x;
    const int wave = t >> 5;
    const int lane = t & 31;
    const int e0   = blockIdx.x * TILE_E;

    if (t < TILE_E) {
        tiSh[t] = tIdx[e0 + t];
        siSh[t] = sIdx[e0 + t];
    }
    __syncthreads();

    const int myRow = t >> 1;            // edge row this thread stages
    const int myTi  = tiSh[myRow];
    const int mySi  = siSh[myRow];
    const int colb  = (t & 1) * 32;      // 32 halves per thread

    v8f acc[8];
#pragma unroll
    for (int n = 0; n < 8; ++n)
#pragma unroll
        for (int r = 0; r < 8; ++r) acc[n][r] = 0.0f;

    const int mrow  = wave * 16 + (lane & 15);
    const int halfK = (lane < 16) ? 0 : 1;

    // ---- W tile: pure f16 copy via async-to-LDS (1024 x 16B per chunk) ----
    auto stageW = [&](int kc, int buf) {
        const _Float16* gsrc = wPack + (size_t)kc * (GATE_W * KC);
        _Float16* lbase = Wsh + buf * (GATE_W * WSTR);
#pragma unroll
        for (int s = 0; s < 4; ++s) {
            int q   = t + s * 256;       // 0..1023 16B-transfers
            int row = q >> 3, seg = q & 7;
            async_load_lds16(gsrc + q * 8, lbase + row * WSTR + seg * 8);
        }
    };

    // ---- A tile: gather f32 (issue early), relu+cvt+store later ----
    float4 fA[8];
    auto issueA = [&](int kc) {
        const int k0 = kc * KC + colb;
        const float* gp = (kc < 8)
            ? (tgtF + (size_t)myTi * FEA + k0)
            : (srcF + (size_t)mySi * FEA + (k0 - FEA));
#pragma unroll
        for (int v = 0; v < 8; ++v) fA[v] = ((const float4*)gp)[v];
    };
    auto commitA = [&](int buf) {
        _Float16* dst = Ash + buf * (TILE_E * ASTR) + myRow * ASTR + colb;
#pragma unroll
        for (int v = 0; v < 4; ++v) {
            float4 f0 = fA[2 * v + 0];
            float4 f1 = fA[2 * v + 1];
            v8h h;
            h[0] = (_Float16)fmaxf(f0.x, 0.0f);
            h[1] = (_Float16)fmaxf(f0.y, 0.0f);
            h[2] = (_Float16)fmaxf(f0.z, 0.0f);
            h[3] = (_Float16)fmaxf(f0.w, 0.0f);
            h[4] = (_Float16)fmaxf(f1.x, 0.0f);
            h[5] = (_Float16)fmaxf(f1.y, 0.0f);
            h[6] = (_Float16)fmaxf(f1.z, 0.0f);
            h[7] = (_Float16)fmaxf(f1.w, 0.0f);
            *(v8h*)(dst + 8 * v) = h;
        }
    };

    // ---- 16 WMMAs on one buffered chunk (B loads software-pipelined) ----
    auto compute = [&](int buf) {
        const _Float16* aB = Ash + buf * (TILE_E * ASTR);
        const _Float16* wB = Wsh + buf * (GATE_W * WSTR);
#pragma unroll
        for (int ks = 0; ks < 2; ++ks) {
            const int abase = ks * 32 + halfK * 8;
            v8h alo = *(const v8h*)(aB + mrow * ASTR + abase);
            v8h ahi = *(const v8h*)(aB + mrow * ASTR + abase + 16);
            v16h afrag = __builtin_shufflevector(alo, ahi,
                0, 1, 2, 3, 4, 5, 6, 7, 8, 9, 10, 11, 12, 13, 14, 15);

            const int kb = ks * 32 + halfK * 8;
            const _Float16* bp = wB + (lane & 15) * WSTR + kb;
            v8h b0 = *(const v8h*)(bp);
            v8h b1 = *(const v8h*)(bp + 16);
#pragma unroll
            for (int n = 0; n < 8; ++n) {
                v16h bfrag = __builtin_shufflevector(b0, b1,
                    0, 1, 2, 3, 4, 5, 6, 7, 8, 9, 10, 11, 12, 13, 14, 15);
                if (n < 7) {   // issue next tile's loads before this WMMA
                    const _Float16* np = bp + (n + 1) * (16 * WSTR);
                    b0 = *(const v8h*)(np);
                    b1 = *(const v8h*)(np + 16);
                }
                acc[n] = __builtin_amdgcn_wmma_f32_16x16x32_f16(
                    false, afrag, false, bfrag, (short)0, acc[n], false, false);
            }
        }
    };

    // ---- pipeline ----
    stageW(0, 0);
    issueA(0);
    commitA(0);
    wait_async0();
    __syncthreads();

    for (int kc = 0; kc < NCHUNK; ++kc) {
        const int cur = kc & 1;
        if (kc < NCHUNK - 1) {           // overlap next chunk's fills w/ compute
            stageW(kc + 1, cur ^ 1);
            issueA(kc + 1);
        }
        compute(cur);
        if (kc < NCHUNK - 1) {
            commitA(cur ^ 1);
            wait_async0();
        }
        __syncthreads();
    }

    // ---- epilogue: gate[e] = mean_n sigmoid(C[e][n] + b[n]) ----
    float bv[8];
#pragma unroll
    for (int n = 0; n < 8; ++n) bv[n] = bvec[n * 16 + (lane & 15)];

#pragma unroll
    for (int r = 0; r < 8; ++r) {
        float s = 0.0f;
#pragma unroll
        for (int n = 0; n < 8; ++n)
            s += sigmoidf_fast(acc[n][r] + bv[n]);
        s += __shfl_xor(s, 1, 32);       // reduce 16 N-lanes per half-wave
        s += __shfl_xor(s, 2, 32);
        s += __shfl_xor(s, 4, 32);
        s += __shfl_xor(s, 8, 32);
        if ((lane & 15) == 0) {
            gateSh[wave * 16 + ((lane < 16) ? 0 : 8) + r] = s * (1.0f / (float)GATE_W);
        }
    }
    __syncthreads();

    // ---- scatter: msg = src_f * gate -> atomic add; count atomics ----
    if (t < TILE_E) atomicAdd(&cnt[tiSh[t]], 1.0f);

    for (int i = t; i < TILE_E * (FEA / 4); i += 256) {
        const int e  = i >> 7;
        const int j4 = i & 127;
        const float g = gateSh[e];
        const float4 sf = ((const float4*)(srcF + (size_t)siSh[e] * FEA))[j4];
        float* ob = out + (size_t)tiSh[e] * FEA + j4 * 4;
        atomicAdd(ob + 0, sf.x * g);
        atomicAdd(ob + 1, sf.y * g);
        atomicAdd(ob + 2, sf.z * g);
        atomicAdd(ob + 3, sf.w * g);
    }
}

// ---------------------------------------------------------------------------
// Normalize: sums / max(count,1)
// ---------------------------------------------------------------------------
__global__ void __launch_bounds__(256) msdn_norm_kernel(float* out, const float* cnt) {
    int i = blockIdx.x * 256 + threadIdx.x;
    if (i < N_NODES * FEA) {
        float c = cnt[i >> 9];           // FEA == 512
        out[i] = out[i] / fmaxf(c, 1.0f);
    }
}

extern "C" void kernel_launch(void* const* d_in, const int* in_sizes, int n_in,
                              void* d_out, int out_size, void* d_ws, size_t ws_size,
                              hipStream_t stream) {
    const float* tgtF = (const float*)d_in[0];
    const float* srcF = (const float*)d_in[1];
    const float* w    = (const float*)d_in[2];
    const float* bvec = (const float*)d_in[3];
    const int*   tIdx = (const int*)d_in[4];
    const int*   sIdx = (const int*)d_in[5];
    float* out = (float*)d_out;

    float*    cnt   = (float*)d_ws;                         // 16 KB
    _Float16* wPack = (_Float16*)((char*)d_ws + 16384);     // 256 KB

    (void)in_sizes; (void)n_in; (void)out_size; (void)ws_size;

    {
        int blocks = (N_NODES * FEA + 255) / 256;           // covers wPack range too
        msdn_prep_kernel<<<blocks, 256, 0, stream>>>(w, out, cnt, wPack);
    }
    {
        int blocks = N_EDGES / TILE_E;                      // 512 blocks x 256 thr
        msdn_edge_kernel<<<blocks, 256, 0, stream>>>(tgtF, srcF, wPack, bvec,
                                                     tIdx, sIdx, out, cnt);
    }
    {
        int blocks = (N_NODES * FEA + 255) / 256;
        msdn_norm_kernel<<<blocks, 256, 0, stream>>>(out, cnt);
    }
}
